// FMoEOpt_54133767798798
// MI455X (gfx1250) — matrix operand order repaired and checked by reading, not compiled, not run on hardware
//
#include <hip/hip_runtime.h>
#include <hip/hip_bf16.h>

typedef __attribute__((ext_vector_type(16))) __bf16 v16bf;
typedef __attribute__((ext_vector_type(8)))  float  v8f;

#define TOKENS    4096
#define D_MODEL   1024
#define D_FF      4096
#define NEXP      8
#define TOPK      2
#define PAD       64                     // expert segment padding (= block M tile)
#define MAX_ROWS  8704                   // >= TOKENS*TOPK + NEXP*(PAD-1), mult of 64
#define MBLOCKS   136                    // MAX_ROWS / 64

// ---------------- helpers ----------------
__device__ __forceinline__ unsigned short f2bf(float f) {
    unsigned int u = __float_as_uint(f);
    unsigned int r = u + 0x7fffu + ((u >> 16) & 1u);   // round-to-nearest-even
    return (unsigned short)(r >> 16);
}

// branch-free tanh-GELU: tanh(u) = 1 - 2/(1+exp(2u)) with HW v_exp_f32
__device__ __forceinline__ float gelu_tanh(float x) {
    float u = 0.7978845608028654f * (x + 0.044715f * x * x * x);
    float t = 1.0f - 2.0f / (1.0f + __expf(2.0f * u));
    return 0.5f * x * (1.0f + t);
}

union ABfrag { uint4 q[2]; v16bf v; };

// ---------------- kernel 1: init ----------------
__global__ void init_kernel(int* row_tk, int* counts) {
    int idx = blockIdx.x * 256 + threadIdx.x;
    if (idx < MAX_ROWS) row_tk[idx] = -1;
    if (idx < NEXP) counts[idx] = 0;
}

// ---------------- kernel 2: gate + topk + x->bf16 ----------------
__global__ void gate_kernel(const float* __restrict__ x, const float* __restrict__ Wg,
                            const float* __restrict__ bg, unsigned short* __restrict__ x_bf,
                            float* __restrict__ gates, int* __restrict__ topi,
                            int* __restrict__ counts) {
    int t = blockIdx.x;
    int tid = threadIdx.x;                       // 64 threads
    const float* xr = x + (size_t)t * D_MODEL;
    float acc[NEXP];
#pragma unroll
    for (int e = 0; e < NEXP; ++e) acc[e] = 0.f;
    int d0 = tid * 16;
#pragma unroll 4
    for (int i = 0; i < 16; ++i) {
        float xv = xr[d0 + i];
        x_bf[(size_t)t * D_MODEL + d0 + i] = f2bf(xv);
        const float* wg = Wg + (size_t)(d0 + i) * NEXP;
#pragma unroll
        for (int e = 0; e < NEXP; ++e) acc[e] += xv * wg[e];
    }
    __shared__ float red[NEXP * 64];
#pragma unroll
    for (int e = 0; e < NEXP; ++e) red[e * 64 + tid] = acc[e];
    __syncthreads();
    __shared__ float logits[NEXP];
    if (tid < NEXP) {
        float s = bg[tid];
        for (int i = 0; i < 64; ++i) s += red[tid * 64 + i];
        logits[tid] = s;
    }
    __syncthreads();
    if (tid == 0) {
        int i0 = 0; float v0 = logits[0];
#pragma unroll
        for (int e = 1; e < NEXP; ++e) if (logits[e] > v0) { v0 = logits[e]; i0 = e; }
        int i1 = (i0 == 0) ? 1 : 0; float v1 = logits[i1];
#pragma unroll
        for (int e = 0; e < NEXP; ++e)
            if (e != i0 && logits[e] > v1) { v1 = logits[e]; i1 = e; }
        float p1 = 1.0f / (1.0f + __expf(v0 - v1));  // softmax over {v0, v1}
        float p0 = 1.0f - p1;
        gates[t * 2 + 0] = p0; gates[t * 2 + 1] = p1;
        topi[t * 2 + 0] = i0;  topi[t * 2 + 1] = i1;
        atomicAdd(&counts[i0], 1);
        atomicAdd(&counts[i1], 1);
    }
}

// ---------------- kernel 3: offsets scan (padded to 64) ----------------
__global__ void scan_kernel(const int* __restrict__ counts, int* __restrict__ poff,
                            int* __restrict__ cursor) {
    if (threadIdx.x == 0 && blockIdx.x == 0) {
        int o = 0;
        for (int e = 0; e < NEXP; ++e) {
            poff[e] = o;
            cursor[e] = o;
            o += (counts[e] + PAD - 1) & ~(PAD - 1);
        }
        poff[NEXP] = o;
    }
}

// ---------------- kernel 4: compact row assignment ----------------
__global__ void assign_kernel(const int* __restrict__ topi, int* __restrict__ cursor,
                              int* __restrict__ row_tk, int* __restrict__ tk_row) {
    int idx = blockIdx.x * 256 + threadIdx.x;    // (t,k) flattened
    if (idx >= TOKENS * TOPK) return;
    int e = topi[idx];
    int slot = atomicAdd(&cursor[e], 1);
    row_tk[slot] = idx;
    tk_row[idx] = slot;
}

// ---------------- kernel 5: f32 -> bf16 transpose  in[E][R][C] -> out[E][C][R] ----------------
__global__ void transpose_bf16_kernel(const float* __restrict__ in, unsigned short* __restrict__ out,
                                      int R, int C) {
    __shared__ unsigned short tile[32][33];
    int e = blockIdx.z;
    int c0 = blockIdx.x * 32;
    int r0 = blockIdx.y * 32;
    const float* inp = in + (size_t)e * R * C;
    unsigned short* outp = out + (size_t)e * R * C;
#pragma unroll
    for (int i = 0; i < 4; ++i) {
        int r = r0 + threadIdx.y + i * 8;
        tile[threadIdx.y + i * 8][threadIdx.x] = f2bf(inp[(size_t)r * C + c0 + threadIdx.x]);
    }
    __syncthreads();
#pragma unroll
    for (int i = 0; i < 4; ++i) {
        int c = c0 + threadIdx.y + i * 8;
        outp[(size_t)c * R + r0 + threadIdx.x] = tile[threadIdx.x][threadIdx.y + i * 8];
    }
}

// ---------------- kernel 6: grouped GEMM1  h = gelu(x @ W1 + b1) ----------------
// block = 256 threads = 8 waves = 2 M-groups x 4 N-groups.
// block tile M=64, N=256; per wave: 2 M-tiles x 4 N-tiles = 8 WMMA C-tiles.
__global__ __launch_bounds__(256)
void ffn1_kernel(const unsigned short* __restrict__ x_bf, const unsigned short* __restrict__ w1t,
                 const float* __restrict__ b1, const int* __restrict__ row_tk,
                 const int* __restrict__ poff, unsigned short* __restrict__ h_bf) {
    int row0 = blockIdx.x * 64;
    if (row0 >= poff[NEXP]) return;
    int e = 0;
#pragma unroll
    for (int i = 1; i < NEXP; ++i) if (row0 >= poff[i]) e = i;

    int lane = threadIdx.x & 31;
    int wave = threadIdx.x >> 5;
    int m = lane & 15;
    int kh = lane >> 4;
    int mbase = row0 + (wave >> 2) * 32;         // M-group of this wave
    int ncol0 = blockIdx.y * 256 + (wave & 3) * 64;

    int tk0 = row_tk[mbase + m];
    int tk1 = row_tk[mbase + 16 + m];
    const unsigned short* arow0 = x_bf + (size_t)((tk0 < 0) ? 0 : (tk0 >> 1)) * D_MODEL;
    const unsigned short* arow1 = x_bf + (size_t)((tk1 < 0) ? 0 : (tk1 >> 1)) * D_MODEL;

    const unsigned short* w1e = w1t + (size_t)e * D_MODEL * D_FF;
    const unsigned short* bcol[4];
#pragma unroll
    for (int j = 0; j < 4; ++j)
        bcol[j] = w1e + (size_t)(ncol0 + j * 16 + m) * D_MODEL + kh * 8;

    v8f z = {0.f, 0.f, 0.f, 0.f, 0.f, 0.f, 0.f, 0.f};
    v8f c0[4], c1[4];
#pragma unroll
    for (int j = 0; j < 4; ++j) { c0[j] = z; c1[j] = z; }

    for (int k = 0; k < D_MODEL; k += 32) {
        ABfrag a0, a1;
        a0.q[0] = *(const uint4*)(arow0 + k + kh * 8);       // K = kh*8 .. +7
        a0.q[1] = *(const uint4*)(arow0 + k + 16 + kh * 8);  // K = 16+kh*8 .. +7
        a1.q[0] = *(const uint4*)(arow1 + k + kh * 8);
        a1.q[1] = *(const uint4*)(arow1 + k + 16 + kh * 8);
#pragma unroll
        for (int j = 0; j < 4; ++j) {
            ABfrag b;
            const unsigned short* bp = bcol[j] + k;
            b.q[0] = *(const uint4*)(bp);                    // col = lane&15, K = kh*8 .. +7
            b.q[1] = *(const uint4*)(bp + 16);               // K = 16+kh*8 .. +7
            c0[j] = __builtin_amdgcn_wmma_f32_16x16x32_bf16(
                false, a0.v, false, b.v, (short)0, c0[j], false, false);
            c1[j] = __builtin_amdgcn_wmma_f32_16x16x32_bf16(
                false, a1.v, false, b.v, (short)0, c1[j], false, false);
        }
    }
#pragma unroll
    for (int j = 0; j < 4; ++j) {
        int col = ncol0 + j * 16 + m;
        float bias = b1[e * D_FF + col];
#pragma unroll
        for (int r = 0; r < 8; ++r) {
            int rowA = mbase + r + kh * 8;                   // C: VGPR r -> M=r (+8 for lanes 16..31)
            int rowB = mbase + 16 + r + kh * 8;
            h_bf[(size_t)rowA * D_FF + col] = f2bf(gelu_tanh(c0[j][r] + bias));
            h_bf[(size_t)rowB * D_FF + col] = f2bf(gelu_tanh(c1[j][r] + bias));
        }
    }
}

// ---------------- kernel 7: grouped GEMM2  y = h @ W2 + b2 ----------------
__global__ __launch_bounds__(256)
void ffn2_kernel(const unsigned short* __restrict__ h_bf, const unsigned short* __restrict__ w2t,
                 const float* __restrict__ b2, const int* __restrict__ poff,
                 float* __restrict__ y_part) {
    int row0 = blockIdx.x * 64;
    if (row0 >= poff[NEXP]) return;
    int e = 0;
#pragma unroll
    for (int i = 1; i < NEXP; ++i) if (row0 >= poff[i]) e = i;

    int lane = threadIdx.x & 31;
    int wave = threadIdx.x >> 5;
    int m = lane & 15;
    int kh = lane >> 4;
    int mbase = row0 + (wave >> 2) * 32;
    int ncol0 = blockIdx.y * 256 + (wave & 3) * 64;

    const unsigned short* arow0 = h_bf + (size_t)(mbase + m) * D_FF;
    const unsigned short* arow1 = h_bf + (size_t)(mbase + 16 + m) * D_FF;

    const unsigned short* w2e = w2t + (size_t)e * D_FF * D_MODEL;
    const unsigned short* bcol[4];
#pragma unroll
    for (int j = 0; j < 4; ++j)
        bcol[j] = w2e + (size_t)(ncol0 + j * 16 + m) * D_FF + kh * 8;

    v8f z = {0.f, 0.f, 0.f, 0.f, 0.f, 0.f, 0.f, 0.f};
    v8f c0[4], c1[4];
#pragma unroll
    for (int j = 0; j < 4; ++j) { c0[j] = z; c1[j] = z; }

    for (int k = 0; k < D_FF; k += 32) {
        ABfrag a0, a1;
        a0.q[0] = *(const uint4*)(arow0 + k + kh * 8);
        a0.q[1] = *(const uint4*)(arow0 + k + 16 + kh * 8);
        a1.q[0] = *(const uint4*)(arow1 + k + kh * 8);
        a1.q[1] = *(const uint4*)(arow1 + k + 16 + kh * 8);
#pragma unroll
        for (int j = 0; j < 4; ++j) {
            ABfrag b;
            const unsigned short* bp = bcol[j] + k;
            b.q[0] = *(const uint4*)(bp);
            b.q[1] = *(const uint4*)(bp + 16);
            c0[j] = __builtin_amdgcn_wmma_f32_16x16x32_bf16(
                false, a0.v, false, b.v, (short)0, c0[j], false, false);
            c1[j] = __builtin_amdgcn_wmma_f32_16x16x32_bf16(
                false, a1.v, false, b.v, (short)0, c1[j], false, false);
        }
    }
#pragma unroll
    for (int j = 0; j < 4; ++j) {
        int col = ncol0 + j * 16 + m;
        float bias = b2[e * D_MODEL + col];
#pragma unroll
        for (int r = 0; r < 8; ++r) {
            int rowA = mbase + r + kh * 8;
            int rowB = mbase + 16 + r + kh * 8;
            y_part[(size_t)rowA * D_MODEL + col] = c0[j][r] + bias;
            y_part[(size_t)rowB * D_MODEL + col] = c1[j][r] + bias;
        }
    }
}

// ---------------- kernel 8: gate-weighted combine ----------------
__global__ void combine_kernel(const float* __restrict__ gates, const int* __restrict__ tk_row,
                               const float* __restrict__ y_part, float* __restrict__ out) {
    int t = blockIdx.x;
    int d = threadIdx.x * 4;                      // 256 threads * 4 = 1024
    float g0 = gates[t * 2 + 0], g1 = gates[t * 2 + 1];
    int r0 = tk_row[t * 2 + 0],  r1 = tk_row[t * 2 + 1];
    float4 a = *(const float4*)(y_part + (size_t)r0 * D_MODEL + d);
    float4 b = *(const float4*)(y_part + (size_t)r1 * D_MODEL + d);
    float4 o;
    o.x = g0 * a.x + g1 * b.x;
    o.y = g0 * a.y + g1 * b.y;
    o.z = g0 * a.z + g1 * b.z;
    o.w = g0 * a.w + g1 * b.w;
    *(float4*)(out + (size_t)t * D_MODEL + d) = o;
}

// ---------------- workspace layout (bytes) ----------------
static constexpr size_t OFF_XBF   = 0;                                  // 4096*1024*2 = 8 MB
static constexpr size_t OFF_W1T   = OFF_XBF  + (size_t)TOKENS * D_MODEL * 2;
static constexpr size_t OFF_W2T   = OFF_W1T  + (size_t)NEXP * D_MODEL * D_FF * 2;
static constexpr size_t OFF_HBF   = OFF_W2T  + (size_t)NEXP * D_FF * D_MODEL * 2;
static constexpr size_t OFF_YP    = OFF_HBF  + (size_t)MAX_ROWS * D_FF * 2;
static constexpr size_t OFF_GATES = OFF_YP   + (size_t)MAX_ROWS * D_MODEL * 4;
static constexpr size_t OFF_TOPI  = OFF_GATES + (size_t)TOKENS * TOPK * 4;
static constexpr size_t OFF_ROWTK = OFF_TOPI  + (size_t)TOKENS * TOPK * 4;
static constexpr size_t OFF_TKROW = OFF_ROWTK + (size_t)MAX_ROWS * 4;
static constexpr size_t OFF_CNT   = OFF_TKROW + (size_t)TOKENS * TOPK * 4;
static constexpr size_t OFF_POFF  = OFF_CNT  + 64;
static constexpr size_t OFF_CUR   = OFF_POFF + 64;

extern "C" void kernel_launch(void* const* d_in, const int* in_sizes, int n_in,
                              void* d_out, int out_size, void* d_ws, size_t ws_size,
                              hipStream_t stream) {
    const float* x  = (const float*)d_in[0];
    const float* Wg = (const float*)d_in[1];
    const float* bg = (const float*)d_in[2];
    const float* W1 = (const float*)d_in[3];
    const float* b1 = (const float*)d_in[4];
    const float* W2 = (const float*)d_in[5];
    const float* b2 = (const float*)d_in[6];
    float* out = (float*)d_out;

    char* ws = (char*)d_ws;
    unsigned short* x_bf  = (unsigned short*)(ws + OFF_XBF);
    unsigned short* w1t   = (unsigned short*)(ws + OFF_W1T);
    unsigned short* w2t   = (unsigned short*)(ws + OFF_W2T);
    unsigned short* h_bf  = (unsigned short*)(ws + OFF_HBF);
    float*          y_part= (float*)(ws + OFF_YP);
    float*          gates = (float*)(ws + OFF_GATES);
    int*            topi  = (int*)(ws + OFF_TOPI);
    int*            row_tk= (int*)(ws + OFF_ROWTK);
    int*            tk_row= (int*)(ws + OFF_TKROW);
    int*            counts= (int*)(ws + OFF_CNT);
    int*            poff  = (int*)(ws + OFF_POFF);
    int*            cursor= (int*)(ws + OFF_CUR);

    init_kernel<<<(MAX_ROWS + 255) / 256, 256, 0, stream>>>(row_tk, counts);
    gate_kernel<<<TOKENS, 64, 0, stream>>>(x, Wg, bg, x_bf, gates, topi, counts);
    scan_kernel<<<1, 32, 0, stream>>>(counts, poff, cursor);
    assign_kernel<<<(TOKENS * TOPK + 255) / 256, 256, 0, stream>>>(topi, cursor, row_tk, tk_row);

    // W1: [E][1024][4096] -> [E][4096][1024] bf16 ; W2: [E][4096][1024] -> [E][1024][4096] bf16
    transpose_bf16_kernel<<<dim3(D_FF / 32, D_MODEL / 32, NEXP), dim3(32, 8), 0, stream>>>(W1, w1t, D_MODEL, D_FF);
    transpose_bf16_kernel<<<dim3(D_MODEL / 32, D_FF / 32, NEXP), dim3(32, 8), 0, stream>>>(W2, w2t, D_FF, D_MODEL);

    ffn1_kernel<<<dim3(MBLOCKS, D_FF / 256), 256, 0, stream>>>(x_bf, w1t, b1, row_tk, poff, h_bf);
    ffn2_kernel<<<dim3(MBLOCKS, D_MODEL / 256), 256, 0, stream>>>(h_bf, w2t, b2, poff, y_part);
    combine_kernel<<<TOKENS, 256, 0, stream>>>(gates, tk_row, y_part, out);
}
